// Attention_layer_44968307589649
// MI455X (gfx1250) — matrix operand order repaired and checked by reading, not compiled, not run on hardware
//
#include <hip/hip_runtime.h>
#include <hip/hip_bf16.h>
#include <math.h>

// Problem sizes (fixed per reference)
#define N_LAB 64
#define B_SZ  1024
#define EMB_  256
#define UEMB_ 256
#define H_SZ  1024
#define KTOT  512   // EMB_ + UEMB_ : bf16 k-extent of the transposed weight
#define KP    264   // padded LDS k-stride (bf16 elems) to dodge bank conflicts

// Flip to 0 if the gfx1250 assembler rejects the async mnemonics.
#define ASYNC_STAGE 1

typedef __attribute__((ext_vector_type(16))) __bf16       v16bf;
typedef __attribute__((ext_vector_type(8)))  float        v8f;
typedef __attribute__((ext_vector_type(4)))  unsigned int v4u;
typedef __attribute__((ext_vector_type(4)))  float        v4f;

union BFrag { v16bf v; v4u u[2]; };

static __device__ inline unsigned short f2bf_bits(float f) {
  union { float f; unsigned u; } c; c.f = f;
  unsigned u = c.u + 0x7FFFu + ((c.u >> 16) & 1u);  // round-to-nearest-even
  return (unsigned short)(u >> 16);
}
static __device__ inline __bf16 f2bf(float f) {
  union { unsigned short s; __bf16 b; } c; c.s = f2bf_bits(f);
  return c.b;
}

// 16-byte Global -> LDS copy on the CDNA5 async path (ASYNCcnt-tracked).
static __device__ inline void async_copy16(unsigned short* lds_dst,
                                           const unsigned short* gsrc) {
#if ASYNC_STAGE
  // LDS byte address = low 32 bits of the generic shared pointer (aperture
  // truncation, ISA 10.2).  Per-lane vdst/vaddr per ISA 15.18.3.
  unsigned int lds_off = (unsigned int)(uintptr_t)lds_dst;
  asm volatile("global_load_async_to_lds_b128 %0, %1, off"
               :
               : "v"(lds_off), "v"(gsrc)
               : "memory");
#else
  *(v4u*)lds_dst = *(const v4u*)gsrc;
#endif
}

static __device__ inline void async_wait() {
#if ASYNC_STAGE
  asm volatile("s_wait_asynccnt 0" ::: "memory");
#endif
}

// Load a full 16x256 A strip (bf16) for this wave into 8 fragments,
// in contiguous k order (same as load_b_frag): lane l holds row M=l%16;
// lanes 0-15 hold K = kb*32+0..15, lanes 16-31 hold kb*32+16..31.
static __device__ inline void load_a_frags(const float* __restrict__ row,
                                           int lane, v16bf a[8]) {
  const int base0 = (lane < 16) ? 0 : 16;
#pragma unroll
  for (int kb = 0; kb < 8; ++kb) {
    const float* p0 = row + kb * 32 + base0;
    v4f x0 = *(const v4f*)(p0);
    v4f x1 = *(const v4f*)(p0 + 4);
    v4f y0 = *(const v4f*)(p0 + 8);
    v4f y1 = *(const v4f*)(p0 + 12);
    v16bf af;
#pragma unroll
    for (int j = 0; j < 4; ++j) {
      af[j]      = f2bf(x0[j]);
      af[4 + j]  = f2bf(x1[j]);
      af[8 + j]  = f2bf(y0[j]);
      af[12 + j] = f2bf(y1[j]);
    }
    a[kb] = af;
  }
}

// Stage a 16(H) x 256(K) bf16 chunk of the pre-transposed weight Wt into
// LDS via async Global->LDS b128 copies: thread t moves 32 contiguous bytes.
static __device__ inline void stage_w_tile(unsigned short* ldsW,
                                           const unsigned short* __restrict__ Wt,
                                           int kOff, int hbase, int tid) {
  const int h   = tid >> 4;         // 0..15
  const int seg = (tid & 15) * 16;  // halves, 32B-aligned
  const unsigned short* g = Wt + (size_t)(hbase + h) * KTOT + kOff + seg;
  unsigned short* l = ldsW + h * KP + seg;
  async_copy16(l, g);
  async_copy16(l + 8, g + 8);
#if !ASYNC_STAGE
  (void)0;
#endif
}

// B fragment (32x16, K x N): lane l holds column N=l%16, 16 contiguous K
// halves starting at kb*32 + (l<16 ? 0 : 16) -> two ds_load_b128.
static __device__ inline v16bf load_b_frag(const unsigned short* ldsW,
                                           int lane, int kb) {
  const int h  = lane & 15;
  const int k0 = kb * 32 + ((lane < 16) ? 0 : 16);
  BFrag f;
  f.u[0] = *(const v4u*)(ldsW + h * KP + k0);
  f.u[1] = *(const v4u*)(ldsW + h * KP + k0 + 8);
  return f.v;
}

// ---- Kernel 0: one-time weight transpose + bf16 convert ----------------
// Wt[h][k] = bf16(W1[k][h]), h in [0,1024), k in [0,512).
__global__ __launch_bounds__(256) void k0_cvtW(const float* __restrict__ W1,
                                               unsigned short* __restrict__ Wt) {
  const int tid   = threadIdx.x;
  const int hbase = blockIdx.x * 16;         // 64 blocks
  const int h     = hbase + (tid & 15);      // 64B-coalesced global reads
  const int kk0   = tid >> 4;                // 0..15
#pragma unroll 4
  for (int p = 0; p < 32; ++p) {
    const int k = kk0 + p * 16;              // covers 0..511
    Wt[(size_t)h * KTOT + k] = f2bf_bits(W1[(size_t)k * H_SZ + h]);
  }
}

// ---- Kernel 1: unl_h[b,h] = unlabel_emb @ W1[EMB:, :] + b1 -------------
__global__ __launch_bounds__(256) void k1_unl(const float* __restrict__ unlabel_emb,
                                              const unsigned short* __restrict__ Wt,
                                              const float* __restrict__ b1,
                                              float* __restrict__ unl_h) {
  __shared__ unsigned short ldsW[16 * KP];
  const int tid  = threadIdx.x;
  const int lane = tid & 31, wave = tid >> 5;
  const int mbase = blockIdx.x * 128 + wave * 16;
  const int hgrp  = blockIdx.y;  // 8 groups of 128 columns

  v16bf a[8];
  load_a_frags(unlabel_emb + (size_t)(mbase + (lane & 15)) * UEMB_, lane, a);

  for (int hc = 0; hc < 8; ++hc) {
    const int hbase = hgrp * 128 + hc * 16;
    __syncthreads();
    stage_w_tile(ldsW, Wt, EMB_, hbase, tid);
    async_wait();
    __syncthreads();

    v8f acc = {};
#pragma unroll
    for (int kb = 0; kb < 8; ++kb) {
      v16bf b = load_b_frag(ldsW, lane, kb);
      acc = __builtin_amdgcn_wmma_f32_16x16x32_bf16(false, a[kb], false, b,
                                                    (short)0, acc, false, false);
    }
    const int hcol = hbase + (lane & 15);
    const float b1v = b1[hcol];
    const int m0 = mbase + ((lane < 16) ? 0 : 8);
#pragma unroll
    for (int r = 0; r < 8; ++r)
      unl_h[(size_t)(m0 + r) * H_SZ + hcol] = acc[r] + b1v;
  }
}

// ---- Kernel 2: fused scores --------------------------------------------
// scores[n,b] = sum_h relu((label_emb[n,b]@W1_lab)[h] + unl_h[b,h]) * W2[h] + b2
// written as out[b,n] = scores[n,b] * dis_lab[b,n]  (pre-softmax)
__global__ __launch_bounds__(256) void k2_scores(const float* __restrict__ label_emb,
                                                 const unsigned short* __restrict__ Wt,
                                                 const float* __restrict__ W2,
                                                 const float* __restrict__ b2,
                                                 const float* __restrict__ dis_lab,
                                                 const float* __restrict__ unl_h,
                                                 float* __restrict__ out) {
  __shared__ unsigned short ldsW[16 * KP];
  const int tid  = threadIdx.x;
  const int lane = tid & 31, wave = tid >> 5;
  const int n     = blockIdx.x;
  const int mbase = blockIdx.y * 128 + wave * 16;

  v16bf a[8];
  load_a_frags(label_emb + ((size_t)n * B_SZ + mbase + (lane & 15)) * EMB_,
               lane, a);

  float s[8] = {0.f, 0.f, 0.f, 0.f, 0.f, 0.f, 0.f, 0.f};

  for (int hc = 0; hc < 64; ++hc) {
    const int hbase = hc * 16;
    __syncthreads();
    stage_w_tile(ldsW, Wt, 0, hbase, tid);
    async_wait();
    __syncthreads();

    v8f acc = {};
#pragma unroll
    for (int kb = 0; kb < 8; ++kb) {
      v16bf b = load_b_frag(ldsW, lane, kb);
      acc = __builtin_amdgcn_wmma_f32_16x16x32_bf16(false, a[kb], false, b,
                                                    (short)0, acc, false, false);
    }
    // epilogue: C layout -> lane holds col N = lane%16, rows r + (lane<16?0:8)
    const int hcol = hbase + (lane & 15);
    const float w2v = W2[hcol];
    const int m0 = mbase + ((lane < 16) ? 0 : 8);
#pragma unroll
    for (int r = 0; r < 8; ++r) {
      float v = acc[r] + unl_h[(size_t)(m0 + r) * H_SZ + hcol];
      v = v > 0.f ? v : 0.f;
      s[r] += v * w2v;
    }
  }

  // reduce the 16 h-columns per row across each 16-lane group
#pragma unroll
  for (int r = 0; r < 8; ++r) {
#pragma unroll
    for (int off = 1; off < 16; off <<= 1)
      s[r] += __shfl_xor(s[r], off, 32);
  }

  if ((lane & 15) == 0) {
    const float b2v = b2[0];
    const int m0 = mbase + ((lane < 16) ? 0 : 8);
#pragma unroll
    for (int r = 0; r < 8; ++r) {
      const int bb = m0 + r;
      out[bb * N_LAB + n] = (s[r] + b2v) * dis_lab[bb * N_LAB + n];
    }
  }
}

// ---- Kernel 3: in-place row softmax over N=64 --------------------------
__global__ __launch_bounds__(256) void k3_softmax(float* __restrict__ out) {
  const int lane = threadIdx.x & 31, wave = threadIdx.x >> 5;
  const int b = blockIdx.x * 8 + wave;
  float v0 = out[b * N_LAB + lane];
  float v1 = out[b * N_LAB + lane + 32];
  float m = fmaxf(v0, v1);
#pragma unroll
  for (int off = 1; off < 32; off <<= 1)
    m = fmaxf(m, __shfl_xor(m, off, 32));
  float e0 = __expf(v0 - m), e1 = __expf(v1 - m);
  float sum = e0 + e1;
#pragma unroll
  for (int off = 1; off < 32; off <<= 1)
    sum += __shfl_xor(sum, off, 32);
  const float inv = 1.0f / sum;
  out[b * N_LAB + lane]      = e0 * inv;
  out[b * N_LAB + lane + 32] = e1 * inv;
}

extern "C" void kernel_launch(void* const* d_in, const int* in_sizes, int n_in,
                              void* d_out, int out_size, void* d_ws, size_t ws_size,
                              hipStream_t stream) {
  (void)in_sizes; (void)n_in; (void)out_size; (void)ws_size;
  const float* unlabel_emb = (const float*)d_in[0];
  const float* label_emb   = (const float*)d_in[1];
  const float* dis_lab     = (const float*)d_in[2];
  const float* W1          = (const float*)d_in[3];
  const float* b1          = (const float*)d_in[4];
  const float* W2          = (const float*)d_in[5];
  const float* b2          = (const float*)d_in[6];
  float* out = (float*)d_out;

  float*          unl_h = (float*)d_ws;                       // 4 MB
  unsigned short* Wt    = (unsigned short*)((char*)d_ws +
                              (size_t)B_SZ * H_SZ * sizeof(float)); // 1 MB

  k0_cvtW  <<<dim3(H_SZ / 16), 256, 0, stream>>>(W1, Wt);
  k1_unl   <<<dim3(8, 8),      256, 0, stream>>>(unlabel_emb, Wt, b1, unl_h);
  k2_scores<<<dim3(N_LAB, 8),  256, 0, stream>>>(label_emb, Wt, W2, b2, dis_lab,
                                                 unl_h, out);
  k3_softmax<<<B_SZ / 8, 256, 0, stream>>>(out);
}
